// BartAttention_44229573214525
// MI455X (gfx1250) — matrix operand order repaired and checked by reading, not compiled, not run on hardware
//
#include <hip/hip_runtime.h>
#include <hip/hip_bf16.h>
#include <math.h>

// ---------------------------------------------------------------------------
// BART attention for MI455X (gfx1250, wave32, WMMA).
//   D=1024, H=16, hd=64, B=2, T=2048.
//   All GEMMs on v_wmma_f32_16x16x32_bf16 (bf16 in, f32 accumulate).
//   Attention is single-pass flash-style (no [T,T] tensor in HBM); the
//   reference softmax is f*exp(s)/sum(f*exp(s)) with no max-subtraction,
//   so plain numerator/denominator accumulation reproduces it.
//   V is produced pre-transposed ([B,H,hd,T]) so the attn*V B-fragments are
//   single 32-byte contiguous loads instead of 16 strided scalar loads.
// ---------------------------------------------------------------------------

typedef __bf16 bf16;
typedef bf16  v16bf __attribute__((ext_vector_type(16)));
typedef bf16  v8bf  __attribute__((ext_vector_type(8)));
typedef float v8f   __attribute__((ext_vector_type(8)));

constexpr int   Bc      = 2;
constexpr int   Tc      = 2048;
constexpr int   Dc      = 1024;
constexpr int   Hc      = 16;
constexpr int   HDc     = 64;
constexpr float SCALING = 0.125f;          // 64^-0.5

__device__ __forceinline__ v16bf mk16(v8bf lo, v8bf hi) {
  return __builtin_shufflevector(lo, hi, 0,1,2,3,4,5,6,7,8,9,10,11,12,13,14,15);
}

__device__ __forceinline__ v8f wmma_bf16(v16bf a, v16bf b, v8f c) {
  // (neg_a, A, neg_b, B, c_mod, C, reuse_a, reuse_b)
  return __builtin_amdgcn_wmma_f32_16x16x32_bf16(false, a, false, b, (short)0, c,
                                                 false, false);
}

// A-fragment (16x32 bf16, M x K): lane holds row M=lane%16.
// Lanes 0-15 hold K = 0..15; lanes 16-31 hold K = 16..31 (matches B)
// -> two contiguous 8-element (16 B) loads per lane.
__device__ __forceinline__ v16bf load_a_frag(const bf16* row, int kb, int half) {
  v8bf lo = *(const v8bf*)(row + kb + half * 16);
  v8bf hi = *(const v8bf*)(row + kb + half * 16 + 8);
  return mk16(lo, hi);
}

// ---------------------------------------------------------------------------
__global__ void f32_to_bf16_kernel(const float* __restrict__ in,
                                   bf16* __restrict__ out, int n) {
  int i = blockIdx.x * blockDim.x + threadIdx.x;
  if (i < n) out[i] = (bf16)in[i];
}

// ---------------------------------------------------------------------------
// Fused Q/K/V projection.  One wave -> one 32(M) x 64(N) tile of one output
// (two A fragments share the four B fragments -> 8 WMMAs per k-step).
// C[m,n] = sum_k hs[m,k] * W[n,k] + b[n];  Q additionally scaled.
// Q,K written [B,H,T,hd];  V written transposed [B,H,hd,T].
__global__ void qkv_proj_kernel(const bf16* __restrict__ hsb,
                                const bf16* __restrict__ wq,
                                const bf16* __restrict__ wk,
                                const bf16* __restrict__ wv,
                                const float* __restrict__ biasq,
                                const float* __restrict__ biask,
                                const float* __restrict__ biasv,
                                bf16* __restrict__ qo,
                                bf16* __restrict__ ko,
                                bf16* __restrict__ vo) {
  const int lane = threadIdx.x & 31;
  const int half = lane >> 4;
  const int lr   = lane & 15;
  const int mbase = blockIdx.x * 32;
  const int nbase = blockIdx.y * 64;
  const int which = blockIdx.z;

  const bf16*  W    = (which == 0) ? wq : (which == 1) ? wk : wv;
  const float* bias = (which == 0) ? biasq : (which == 1) ? biask : biasv;
  bf16*        out  = (which == 0) ? qo : (which == 1) ? ko : vo;
  const float  scale = (which == 0) ? SCALING : 1.0f;

  const bf16* arow0 = hsb + (size_t)(mbase + lr) * Dc;
  const bf16* arow1 = hsb + (size_t)(mbase + 16 + lr) * Dc;
  v8f acc[2][4] = {};

  for (int kb = 0; kb < Dc; kb += 32) {
    v16bf a0 = load_a_frag(arow0, kb, half);
    v16bf a1 = load_a_frag(arow1, kb, half);
#pragma unroll
    for (int nt = 0; nt < 4; ++nt) {
      // B-fragment (32x16, K x N): lane holds column N = nbase+nt*16+lr,
      // elements are 16 contiguous K values starting at kb + half*16.
      const bf16* wrow = W + (size_t)(nbase + nt * 16 + lr) * Dc;
      v16bf bf = *(const v16bf*)(wrow + kb + half * 16);
      acc[0][nt] = wmma_bf16(a0, bf, acc[0][nt]);
      acc[1][nt] = wmma_bf16(a1, bf, acc[1][nt]);
    }
  }

#pragma unroll
  for (int nt = 0; nt < 4; ++nt) {
    const int   n  = nbase + nt * 16 + lr;
    const float bv = bias[n];
    const int   h  = n >> 6;        // head
    const int   dp = n & (HDc - 1); // position within head
#pragma unroll
    for (int mt = 0; mt < 2; ++mt) {
#pragma unroll
      for (int r = 0; r < 8; ++r) {
        const int m  = mbase + mt * 16 + r + 8 * half; // C rows: lanes>=16 +8
        const int bi = m >> 11;                        // T = 2048
        const int t  = m & (Tc - 1);
        const bf16 val = (bf16)((acc[mt][nt][r] + bv) * scale);
        if (which == 2)  // V transposed: [B,H,hd,T]
          out[((size_t)(bi * Hc + h) * HDc + dp) * Tc + t] = val;
        else             // Q,K: [B,H,T,hd]
          out[(((size_t)bi * Hc + h) * Tc + t) * HDc + dp] = val;
      }
    }
  }
}

// ---------------------------------------------------------------------------
// Flash-style attention.  One wave owns one (b,h,16-row t-tile); streams the
// s dimension in 32-wide chunks.  P = f * exp(QK^T) round-trips through a
// per-wave LDS tile to convert from C-fragment layout to A-fragment layout.
__global__ void attn_kernel(const bf16* __restrict__ qb,
                            const bf16* __restrict__ kb,
                            const bf16* __restrict__ vtb,
                            const float* __restrict__ F,
                            bf16* __restrict__ aob) {
  __shared__ __align__(16) bf16 lds[4][16 * 40];   // per-wave 16 x 32 P, pad->40
  const int lane = threadIdx.x & 31;
  const int wid  = threadIdx.x >> 5;
  const int half = lane >> 4;
  const int lr   = lane & 15;

  const int tile  = blockIdx.x * 4 + wid;    // 4096 tiles total
  const int bh    = tile >> 7;               // T/16 = 128 tiles per (b,h)
  const int tbase = (tile & 127) << 4;
  const int b     = bh >> 4;
  const int h     = bh & 15;

  const bf16* Q   = qb  + (size_t)bh * Tc * HDc;
  const bf16* K   = kb  + (size_t)bh * Tc * HDc;
  const bf16* Vt  = vtb + (size_t)bh * HDc * Tc;   // [hd, T]
  const float* Fb = F + (size_t)b * Tc * Tc;

  // Q A-fragments: rows t = tbase+lr, hd split into two K=32 chunks.
  const bf16* qrow = Q + (size_t)(tbase + lr) * HDc;
  const v16bf aq0 = load_a_frag(qrow, 0, half);
  const v16bf aq1 = load_a_frag(qrow, 32, half);

  v8f   o[4]    = {};
  float dacc[8] = {0.f, 0.f, 0.f, 0.f, 0.f, 0.f, 0.f, 0.f};
  bf16* P = lds[wid];

  for (int sb = 0; sb < Tc; sb += 32) {
#pragma unroll
    for (int j = 0; j < 2; ++j) {
      const int   scol = sb + j * 16 + lr;
      const bf16* krow = K + (size_t)scol * HDc;       // B column s = row of K
      const v16bf bk0  = *(const v16bf*)(krow + half * 16);
      const v16bf bk1  = *(const v16bf*)(krow + 32 + half * 16);
      v8f c = {};
      c = wmma_bf16(aq0, bk0, c);
      c = wmma_bf16(aq1, bk1, c);
      // P = f * exp(S); accumulate row denominators; stage P^bf16 in LDS.
#pragma unroll
      for (int r = 0; r < 8; ++r) {
        const int tg = tbase + r + 8 * half;
        const float p = Fb[(size_t)tg * Tc + scol] * __expf(c[r]);
        dacc[r] += p;
        P[(r + 8 * half) * 40 + j * 16 + lr] = (bf16)p;
      }
    }
    // Reload P as an A-fragment (16 x 32): row t = lr, contiguous s runs.
    const bf16* prow = P + lr * 40;
    const v16bf ap = mk16(*(const v8bf*)(prow + half * 16),
                          *(const v8bf*)(prow + half * 16 + 8));
    // V B-fragments from the transposed layout: lane = column d, elements are
    // 16 contiguous s values -> one aligned 32-byte load per d-tile.
#pragma unroll
    for (int dt = 0; dt < 4; ++dt) {
      const v16bf bvf =
          *(const v16bf*)(Vt + (size_t)(dt * 16 + lr) * Tc + sb + half * 16);
      o[dt] = wmma_bf16(ap, bvf, o[dt]);
    }
  }

  // Denominator: C-fragment rows live in 16-lane halves -> xor-reduce halves.
#pragma unroll
  for (int r = 0; r < 8; ++r) {
    float d = dacc[r];
    d += __shfl_xor(d, 1, 32);
    d += __shfl_xor(d, 2, 32);
    d += __shfl_xor(d, 4, 32);
    d += __shfl_xor(d, 8, 32);
    dacc[r] = 1.0f / d;
  }

  // Store normalized output, heads re-interleaved to [B*T, D] bf16.
#pragma unroll
  for (int dt = 0; dt < 4; ++dt) {
    const int col = h * HDc + dt * 16 + lr;
#pragma unroll
    for (int r = 0; r < 8; ++r) {
      const int tg = tbase + r + 8 * half;
      aob[((size_t)b * Tc + tg) * Dc + col] = (bf16)(o[dt][r] * dacc[r]);
    }
  }
}

// ---------------------------------------------------------------------------
// Output projection: out = attn_out @ Wo^T + bo (f32).  32x64 tile per wave.
__global__ void out_proj_kernel(const bf16* __restrict__ aob,
                                const bf16* __restrict__ wo,
                                const float* __restrict__ biaso,
                                float* __restrict__ out) {
  const int lane = threadIdx.x & 31;
  const int half = lane >> 4;
  const int lr   = lane & 15;
  const int mbase = blockIdx.x * 32;
  const int nbase = blockIdx.y * 64;

  const bf16* arow0 = aob + (size_t)(mbase + lr) * Dc;
  const bf16* arow1 = aob + (size_t)(mbase + 16 + lr) * Dc;
  v8f acc[2][4] = {};

  for (int kb = 0; kb < Dc; kb += 32) {
    v16bf a0 = load_a_frag(arow0, kb, half);
    v16bf a1 = load_a_frag(arow1, kb, half);
#pragma unroll
    for (int nt = 0; nt < 4; ++nt) {
      const bf16* wrow = wo + (size_t)(nbase + nt * 16 + lr) * Dc;
      v16bf bf = *(const v16bf*)(wrow + kb + half * 16);
      acc[0][nt] = wmma_bf16(a0, bf, acc[0][nt]);
      acc[1][nt] = wmma_bf16(a1, bf, acc[1][nt]);
    }
  }

#pragma unroll
  for (int nt = 0; nt < 4; ++nt) {
    const int   n  = nbase + nt * 16 + lr;
    const float bv = biaso[n];
#pragma unroll
    for (int mt = 0; mt < 2; ++mt) {
#pragma unroll
      for (int r = 0; r < 8; ++r) {
        const int m = mbase + mt * 16 + r + 8 * half;
        out[(size_t)m * Dc + n] = acc[mt][nt][r] + bv;
      }
    }
  }
}

// ---------------------------------------------------------------------------
extern "C" void kernel_launch(void* const* d_in, const int* in_sizes, int n_in,
                              void* d_out, int out_size, void* d_ws, size_t ws_size,
                              hipStream_t stream) {
  const float* hs = (const float*)d_in[0];
  const float* F  = (const float*)d_in[1];
  const float* Wq = (const float*)d_in[2];
  const float* bq = (const float*)d_in[3];
  const float* Wk = (const float*)d_in[4];
  const float* bk = (const float*)d_in[5];
  const float* Wv = (const float*)d_in[6];
  const float* bv = (const float*)d_in[7];
  const float* Wo = (const float*)d_in[8];
  const float* bo = (const float*)d_in[9];
  float* out = (float*)d_out;

  const size_t BT = (size_t)Bc * Tc;          // 4096
  const size_t DD = (size_t)Dc * Dc;          // 1M

  // Workspace carve (~48 MiB total), 256-B aligned slices.
  char* p = (char*)d_ws;
  auto carve = [&](size_t elems) {
    bf16* r = (bf16*)p;
    p += (elems * sizeof(bf16) + 255) & ~(size_t)255;
    return r;
  };
  bf16* hsb = carve(BT * Dc);
  bf16* wqb = carve(DD);
  bf16* wkb = carve(DD);
  bf16* wvb = carve(DD);
  bf16* wob = carve(DD);
  bf16* qb  = carve(BT * Dc);   // [B,H,T,hd]
  bf16* kbm = carve(BT * Dc);   // [B,H,T,hd]
  bf16* vtm = carve(BT * Dc);   // [B,H,hd,T]  (transposed)
  bf16* aob = carve(BT * Dc);   // attention output, [B*T, D]

  // 1) f32 -> bf16 conversions.
  {
    const int thr = 256;
    int n_hs = (int)(BT * Dc);
    int n_w  = (int)DD;
    f32_to_bf16_kernel<<<(n_hs + thr - 1) / thr, thr, 0, stream>>>(hs, hsb, n_hs);
    f32_to_bf16_kernel<<<(n_w + thr - 1) / thr, thr, 0, stream>>>(Wq, wqb, n_w);
    f32_to_bf16_kernel<<<(n_w + thr - 1) / thr, thr, 0, stream>>>(Wk, wkb, n_w);
    f32_to_bf16_kernel<<<(n_w + thr - 1) / thr, thr, 0, stream>>>(Wv, wvb, n_w);
    f32_to_bf16_kernel<<<(n_w + thr - 1) / thr, thr, 0, stream>>>(Wo, wob, n_w);
  }

  // 2) Fused QKV projection: grid (M/32, N/64, 3), one wave per 32x64 tile.
  qkv_proj_kernel<<<dim3(BT / 32, Dc / 64, 3), 32, 0, stream>>>(
      hsb, wqb, wkb, wvb, bq, bk, bv, qb, kbm, vtm);

  // 3) Flash attention: 4096 t-tiles, 4 waves per block.
  attn_kernel<<<dim3((Bc * Hc * (Tc / 16)) / 4), 128, 0, stream>>>(
      qb, kbm, vtm, F, aob);

  // 4) Output projection -> f32 d_out.
  out_proj_kernel<<<dim3(BT / 32, Dc / 64), 32, 0, stream>>>(aob, wob, bo, out);
}
